// Static_Backbone_84645215470244
// MI455X (gfx1250) — matrix operand order repaired and checked by reading, not compiled, not run on hardware
//
#include <hip/hip_runtime.h>
#include <hip/hip_bf16.h>

#define NPTS 65536
#define KTAP 27
#define KNNK 16

// Toggle: async-DMA gather (global_load_async_to_lds_b128, ASYNCcnt) vs
// classic VMEM->VGPR->ds_store staging.
#define USE_ASYNC_GATHER 1

typedef __attribute__((ext_vector_type(16))) __bf16 v16bf;
typedef __attribute__((ext_vector_type(8)))  float  v8f;

__device__ __forceinline__ unsigned short f2b(float f) {
  unsigned u = __float_as_uint(f);
  unsigned r = u + 0x7FFFu + ((u >> 16) & 1u);   // round-to-nearest-even
  return (unsigned short)(r >> 16);
}
__device__ __forceinline__ float b2f(unsigned short h) {
  return __uint_as_float(((unsigned)h) << 16);
}

// ---------------------------------------------------------------------------
// Pack f32 weights [K][Cin][Cout] -> bf16 WMMA B-fragments:
// out[k][ci][co][lane][e], lane<16: N=lane,    K=e      (within 32-chunk ci)
//                          lane>=16: N=lane-16, K=16+e
// ---------------------------------------------------------------------------
__global__ void pack_w_kernel(const float* __restrict__ W,
                              unsigned short* __restrict__ out,
                              int K, int Cin, int Cout) {
  const int nci = Cin >> 5, nco = Cout >> 4;
  const size_t total = (size_t)K * nci * nco * 512;
  for (size_t i = (size_t)blockIdx.x * blockDim.x + threadIdx.x; i < total;
       i += (size_t)gridDim.x * blockDim.x) {
    int e    = (int)(i & 15);
    int lane = (int)((i >> 4) & 31);
    size_t f = i >> 9;
    int co = (int)(f % nco);  f /= nco;
    int ci = (int)(f % nci);
    int k  = (int)(f / nci);
    int n  = lane & 15;
    int kk = ((lane >> 4) << 4) + e;               // 0..15 or 16..31
    float w = W[((size_t)k * Cin + (ci << 5) + kk) * Cout + (co << 4) + n];
    out[i] = f2b(w);
  }
}

// zero 128 ushorts (pad page for async gather)
__global__ void zero_page_kernel(unsigned short* __restrict__ zp) {
  if (threadIdx.x < 128) zp[threadIdx.x] = 0;
}

// ---------------------------------------------------------------------------
// Input conv: feats [N,1] f32, 27 taps, W_in [27,1,128] -> bf16 [N,128]
// ---------------------------------------------------------------------------
__global__ __launch_bounds__(256) void in_conv_kernel(
    const float* __restrict__ feats, const int* __restrict__ nbr,
    const float* __restrict__ Win, const float* __restrict__ bin,
    unsigned short* __restrict__ out) {
  int t = blockIdx.x * 256 + threadIdx.x;
  int n = t >> 7;
  int c = t & 127;
  float acc = bin[c];
  #pragma unroll
  for (int k = 0; k < KTAP; ++k) {
    int iv = nbr[n * KTAP + k];
    float f = ((unsigned)iv < (unsigned)NPTS) ? feats[iv] : 0.0f;
    acc += f * Win[k * 128 + c];
  }
  out[(size_t)n * 128 + c] = f2b(acc);
}

// ---------------------------------------------------------------------------
// Templated gathered GEMM via WMMA bf16.
//   Y[n, oco + :COUT] = epilog( sum_k sum_c X[idx[n,k], :CIN] * W[k] )
//   KOFF == 1 -> identity gather (dense 1x1 GEMM).
//   flags: bit0 relu before residual, bit1 relu after residual, bit2 f32 out
// Block: 32 points (two 16-row A tiles), (COUT/16) waves.
// Staging: per-lane async DMA global->LDS (ASYNCcnt), double-buffered across
// taps; pad rows are redirected to a zeroed page.
// ---------------------------------------------------------------------------
template <int CIN, int COUT, int KOFF>
__global__ __launch_bounds__(256) void spconv_t(
    const unsigned short* __restrict__ X,
    const int* __restrict__ idx,
    const unsigned short* __restrict__ Wpk,     // packed [KOFF][NCI][NCO][512]
    const float* __restrict__ bias,             // [COUT] or nullptr
    const unsigned short* __restrict__ resid, int ldr, int rco,
    unsigned short* __restrict__ Ybf, float* __restrict__ Yf,
    int ldy, int oco, int flags,
    const unsigned short* __restrict__ zp) {    // 128-ushort zero page
  constexpr int NCI     = CIN / 32;
  constexpr int NCO     = COUT / 16;
  constexpr int THREADS = NCO * 32;
  constexpr int ROWCH   = CIN / 8;              // 16B chunks per row
  constexpr int CHUNKS  = 32 * ROWCH;           // per 32-row tile
  constexpr int ITERS   = CHUNKS / THREADS;     // integral for all uses

  __shared__ unsigned short lA[2][32 * CIN];

  const int n0   = blockIdx.x << 5;             // 32 points per block
  const int wave = threadIdx.x >> 5;
  const int lane = threadIdx.x & 31;

  auto stage = [&](int k, int buf) {
    #pragma unroll
    for (int it = 0; it < ITERS; ++it) {
      int t  = it * THREADS + threadIdx.x;
      int p  = t / ROWCH;
      int cb = (t - p * ROWCH) << 3;            // half offset in row
      int src = (KOFF > 1) ? idx[(n0 + p) * KOFF + k] : (n0 + p);
#if USE_ASYNC_GATHER
      const unsigned short* gp =
          ((unsigned)src < (unsigned)NPTS) ? (X + (size_t)src * CIN + cb)
                                           : (zp + cb);
      unsigned long long ga = (unsigned long long)(size_t)gp;
      unsigned lo = (unsigned)(size_t)(&lA[buf][p * CIN + cb]);
      asm volatile("global_load_async_to_lds_b128 %0, %1, off"
                   :: "v"(lo), "v"(ga) : "memory");
#else
      uint4 val;
      if ((unsigned)src < (unsigned)NPTS) {
        val = *(const uint4*)(X + (size_t)src * CIN + cb);
      } else {
        val.x = 0u; val.y = 0u; val.z = 0u; val.w = 0u;
      }
      *(uint4*)(&lA[buf][p * CIN + cb]) = val;
#endif
    }
  };

  v8f acc0 = {}, acc1 = {};
  stage(0, 0);
  const int row = lane & 15;
  const int hi  = (lane >> 4) << 4;             // 0 or 16 halves

  for (int k = 0; k < KOFF; ++k) {
#if USE_ASYNC_GATHER
    // this wave's async DMA into LDS must land before the barrier publishes
    asm volatile("s_wait_asynccnt 0x0" ::: "memory");
#endif
    __syncthreads();
    if (KOFF > 1 && k + 1 < KOFF) {
      stage(k + 1, (k + 1) & 1);
      __builtin_prefetch(
          Wpk + ((size_t)(k + 1) * NCI * NCO + wave) * 512 + (lane << 4), 0, 1);
    }
    const unsigned short* lb = &lA[k & 1][0];
    #pragma unroll
    for (int ci = 0; ci < NCI; ++ci) {
      union { v16bf v; uint4 u[2]; } b, a0, a1;
      const unsigned short* bp =
          Wpk + (((size_t)k * NCI + ci) * NCO + wave) * 512 + (lane << 4);
      b.u[0] = *(const uint4*)(bp);
      b.u[1] = *(const uint4*)(bp + 8);
      const unsigned short* ap0 = lb + row * CIN + (ci << 5) + hi;
      a0.u[0] = *(const uint4*)(ap0);           // K 0..7  (or 16..23)
      a0.u[1] = *(const uint4*)(ap0 + 8);       // K 8..15 (or 24..31)
      const unsigned short* ap1 = ap0 + 16 * CIN;
      a1.u[0] = *(const uint4*)(ap1);
      a1.u[1] = *(const uint4*)(ap1 + 8);
      acc0 = __builtin_amdgcn_wmma_f32_16x16x32_bf16(
          false, a0.v, false, b.v, (short)0, acc0, false, false);
      acc1 = __builtin_amdgcn_wmma_f32_16x16x32_bf16(
          false, a1.v, false, b.v, (short)0, acc1, false, false);
    }
  }

  // epilog: C/D layout: VGPR v, lane -> row v + (lane<16?0:8), col lane&15
  const int col = lane & 15;
  const int mhi = (lane >> 4) << 3;
  const int cg  = (wave << 4) + col;
  float bv = bias ? bias[cg] : 0.0f;
  #pragma unroll
  for (int tile = 0; tile < 2; ++tile) {
    v8f acc = tile ? acc1 : acc0;
    #pragma unroll
    for (int v = 0; v < 8; ++v) {
      int m = n0 + tile * 16 + mhi + v;
      float t = acc[v] + bv;
      if (flags & 1) t = fmaxf(t, 0.0f);
      if (resid) t += b2f(resid[(size_t)m * ldr + rco + cg]);
      if (flags & 2) t = fmaxf(t, 0.0f);
      size_t o = (size_t)m * ldy + oco + cg;
      if (flags & 4) Yf[o] = t;
      else           Ybf[o] = f2b(t);
    }
  }
}

// ---------------------------------------------------------------------------
// KNN attention aggregation: one wave32 per point, 4 channels per lane.
// ---------------------------------------------------------------------------
__global__ __launch_bounds__(256) void knn_attn_agg_kernel(
    const unsigned short* __restrict__ q, const unsigned short* __restrict__ kf,
    const unsigned short* __restrict__ vf, const int* __restrict__ knn,
    unsigned short* __restrict__ agg) {
  const int wid  = blockIdx.x * (blockDim.x >> 5) + (threadIdx.x >> 5);
  const int lane = threadIdx.x & 31;
  const int cb = lane << 2;                     // 4 channels / lane
  float qv[4];
  #pragma unroll
  for (int i = 0; i < 4; ++i) qv[i] = b2f(q[(size_t)wid * 128 + cb + i]);

  int   nb[KNNK];
  float lg[KNNK];
  #pragma unroll
  for (int j = 0; j < KNNK; ++j) {
    int m = knn[wid * KNNK + j];
    nb[j] = m;
    const unsigned short* kp = kf + (size_t)m * 128 + cb;
    float p = qv[0] * b2f(kp[0]) + qv[1] * b2f(kp[1]) +
              qv[2] * b2f(kp[2]) + qv[3] * b2f(kp[3]);
    p += __shfl_xor(p, 16, 32);
    p += __shfl_xor(p,  8, 32);
    p += __shfl_xor(p,  4, 32);
    p += __shfl_xor(p,  2, 32);
    p += __shfl_xor(p,  1, 32);
    lg[j] = p * 0.088388347648318447f;          // 128^-0.5
  }
  float mx = lg[0];
  #pragma unroll
  for (int j = 1; j < KNNK; ++j) mx = fmaxf(mx, lg[j]);
  float s = 0.0f;
  #pragma unroll
  for (int j = 0; j < KNNK; ++j) { lg[j] = __expf(lg[j] - mx); s += lg[j]; }
  float inv = 1.0f / s;
  float a0 = 0, a1 = 0, a2 = 0, a3 = 0;
  #pragma unroll
  for (int j = 0; j < KNNK; ++j) {
    float w = lg[j] * inv;
    const unsigned short* vp = vf + (size_t)nb[j] * 128 + cb;
    a0 += w * b2f(vp[0]); a1 += w * b2f(vp[1]);
    a2 += w * b2f(vp[2]); a3 += w * b2f(vp[3]);
  }
  unsigned short* op = agg + (size_t)wid * 128 + cb;
  op[0] = f2b(a0); op[1] = f2b(a1); op[2] = f2b(a2); op[3] = f2b(a3);
}

// ---------------------------------------------------------------------------
// Host orchestration
// ---------------------------------------------------------------------------
static inline size_t pk_elems(int K, int Cin, int Cout) {
  return (size_t)K * (Cin / 32) * (Cout / 16) * 512;
}

static void pack_w(hipStream_t s, const float* W, unsigned short* out,
                   int K, int Cin, int Cout) {
  pack_w_kernel<<<dim3(1024), dim3(256), 0, s>>>(W, out, K, Cin, Cout);
}

template <int CIN, int COUT, int KOFF>
static void spconv(hipStream_t s, const unsigned short* X, const int* idx,
                   const unsigned short* Wpk, const float* bias,
                   const unsigned short* resid, int ldr, int rco,
                   unsigned short* Ybf, float* Yf, int ldy, int oco,
                   int flags, const unsigned short* zp) {
  dim3 grid(NPTS / 32);
  dim3 block((COUT / 16) * 32);
  spconv_t<CIN, COUT, KOFF><<<grid, block, 0, s>>>(
      X, idx, Wpk, bias, resid, ldr, rco, Ybf, Yf, ldy, oco, flags, zp);
}

extern "C" void kernel_launch(void* const* d_in, const int* in_sizes, int n_in,
                              void* d_out, int out_size, void* d_ws,
                              size_t ws_size, hipStream_t stream) {
  (void)in_sizes; (void)n_in; (void)out_size; (void)ws_size;

  const float* feats   = (const float*)d_in[0];
  const int*   nbr_idx = (const int*)d_in[1];
  const int*   knn_idx = (const int*)d_in[2];
  const float* W_in    = (const float*)d_in[3];
  const float* b_in    = (const float*)d_in[4];
  const float* W_conv0 = (const float*)d_in[5];
  const float* b_conv0 = (const float*)d_in[6];
  const float* W_conv1 = (const float*)d_in[7];
  const float* b_conv1 = (const float*)d_in[8];
  const float* Wq      = (const float*)d_in[9];
  const float* Wk      = (const float*)d_in[10];
  const float* Wv      = (const float*)d_in[11];
  const float* Wo      = (const float*)d_in[12];
  const float* bo      = (const float*)d_in[13];
  const float* W_a0    = (const float*)d_in[14];
  const float* b_a0    = (const float*)d_in[15];
  const float* W_a1    = (const float*)d_in[16];
  const float* b_a1    = (const float*)d_in[17];
  const float* W_a2    = (const float*)d_in[18];
  const float* b_a2    = (const float*)d_in[19];
  const float* W_b0    = (const float*)d_in[20];
  const float* b_b0    = (const float*)d_in[21];
  const float* W_b1    = (const float*)d_in[22];
  const float* b_b1    = (const float*)d_in[23];
  const float* W_out   = (const float*)d_in[24];
  const float* b_out   = (const float*)d_in[25];

  // ---- bump allocator over d_ws ----
  char* base = (char*)d_ws;
  size_t off = 0;
  auto alloc = [&](size_t bytes) -> void* {
    off = (off + 255) & ~(size_t)255;
    void* p = base + off;
    off += bytes;
    return p;
  };

  // zero page for async-gather padding
  unsigned short* zp = (unsigned short*)alloc(128 * 2);

  // packed weights (bf16)
  unsigned short* pk_c0  = (unsigned short*)alloc(pk_elems(3*KTAP,128,128)*2);
  unsigned short* pk_c1  = (unsigned short*)alloc(pk_elems(3*KTAP,128,128)*2);
  unsigned short* pk_out = (unsigned short*)alloc(pk_elems(KTAP,128,128)*2);
  unsigned short* pk_q   = (unsigned short*)alloc(pk_elems(3,128,128)*2);
  unsigned short* pk_k   = (unsigned short*)alloc(pk_elems(3,128,128)*2);
  unsigned short* pk_v   = (unsigned short*)alloc(pk_elems(3,128,128)*2);
  unsigned short* pk_o   = (unsigned short*)alloc(pk_elems(3,128,128)*2);
  unsigned short* pk_a0  = (unsigned short*)alloc(pk_elems(3,128,32)*2);
  unsigned short* pk_a1  = (unsigned short*)alloc(pk_elems(3*KTAP,32,32)*2);
  unsigned short* pk_a2  = (unsigned short*)alloc(pk_elems(3,32,64)*2);
  unsigned short* pk_b0  = (unsigned short*)alloc(pk_elems(3*KTAP,128,32)*2);
  unsigned short* pk_b1  = (unsigned short*)alloc(pk_elems(3*KTAP,32,64)*2);

  // feature buffers (bf16)
  unsigned short* X[6];
  for (int i = 0; i < 6; ++i)
    X[i] = (unsigned short*)alloc((size_t)NPTS * 128 * 2);
  unsigned short* G0 = (unsigned short*)alloc((size_t)NPTS * 32 * 2);
  unsigned short* G1 = (unsigned short*)alloc((size_t)NPTS * 32 * 2);

  // ---- init zero page + pack all weights ----
  zero_page_kernel<<<dim3(1), dim3(128), 0, stream>>>(zp);
  pack_w(stream, W_conv0, pk_c0, 3*KTAP, 128, 128);
  pack_w(stream, W_conv1, pk_c1, 3*KTAP, 128, 128);
  pack_w(stream, W_out,  pk_out, KTAP,   128, 128);
  pack_w(stream, Wq, pk_q, 3, 128, 128);
  pack_w(stream, Wk, pk_k, 3, 128, 128);
  pack_w(stream, Wv, pk_v, 3, 128, 128);
  pack_w(stream, Wo, pk_o, 3, 128, 128);
  pack_w(stream, W_a0, pk_a0, 3, 128, 32);
  pack_w(stream, W_a1, pk_a1, 3*KTAP, 32, 32);
  pack_w(stream, W_a2, pk_a2, 3, 32, 64);
  pack_w(stream, W_b0, pk_b0, 3*KTAP, 128, 32);
  pack_w(stream, W_b1, pk_b1, 3*KTAP, 32, 64);

  // ---- input conv: feats [N,1] -> X[0] [N,128] bf16 ----
  in_conv_kernel<<<dim3(NPTS * 128 / 256), dim3(256), 0, stream>>>(
      feats, nbr_idx, W_in, b_in, X[0]);

  int cur = 0;
  for (int s = 0; s < 3; ++s) {
    int t[5], w = 0;
    for (int i = 0; i < 6; ++i) if (i != cur) t[w++] = i;
    unsigned short* h  = X[t[0]];
    unsigned short* qb = X[t[1]];
    unsigned short* kb = X[t[2]];
    unsigned short* vb = X[t[3]];
    unsigned short* ab = X[t[4]];

    const size_t pkc = pk_elems(KTAP, 128, 128);
    const size_t pkg = pk_elems(1, 128, 128);

    // h = sparse_conv(cur, W_conv0[s]) + b
    spconv<128,128,KTAP>(stream, X[cur], nbr_idx, pk_c0 + s * pkc,
                         b_conv0 + s * 128, nullptr, 0, 0,
                         h, nullptr, 128, 0, 0, zp);
    // q,k,v = h @ Wq/Wk/Wv (no bias)
    spconv<128,128,1>(stream, h, nullptr, pk_q + s * pkg, nullptr,
                      nullptr, 0, 0, qb, nullptr, 128, 0, 0, zp);
    spconv<128,128,1>(stream, h, nullptr, pk_k + s * pkg, nullptr,
                      nullptr, 0, 0, kb, nullptr, 128, 0, 0, zp);
    spconv<128,128,1>(stream, h, nullptr, pk_v + s * pkg, nullptr,
                      nullptr, 0, 0, vb, nullptr, 128, 0, 0, zp);
    // agg = attention(q,k,v)
    knn_attn_agg_kernel<<<dim3(NPTS / 8), dim3(256), 0, stream>>>(
        qb, kb, vb, knn_idx, ab);
    // hA = relu(h + agg @ Wo + bo) -> q buffer
    spconv<128,128,1>(stream, ab, nullptr, pk_o + s * pkg, bo + s * 128,
                      h, 128, 0, qb, nullptr, 128, 0, 2, zp);
    // x = sparse_conv(hA, W_conv1[s]) + b -> k buffer
    spconv<128,128,KTAP>(stream, qb, nbr_idx, pk_c1 + s * pkc,
                         b_conv1 + s * 128, nullptr, 0, 0,
                         kb, nullptr, 128, 0, 0, zp);
    unsigned short* x = kb;
    // irn:
    spconv<128,32,1>(stream, x, nullptr, pk_a0 + s * pk_elems(1,128,32),
                     b_a0 + s * 32, nullptr, 0, 0, G0, nullptr, 32, 0, 1, zp);
    spconv<32,32,KTAP>(stream, G0, nbr_idx, pk_a1 + s * pk_elems(KTAP,32,32),
                       b_a1 + s * 32, nullptr, 0, 0, G1, nullptr, 32, 0, 1, zp);
    spconv<32,64,1>(stream, G1, nullptr, pk_a2 + s * pk_elems(1,32,64),
                    b_a2 + s * 64, x, 128, 0, vb, nullptr, 128, 0, 1, zp);
    spconv<128,32,KTAP>(stream, x, nbr_idx, pk_b0 + s * pk_elems(KTAP,128,32),
                        b_b0 + s * 32, nullptr, 0, 0, G0, nullptr, 32, 0, 1, zp);
    spconv<32,64,KTAP>(stream, G0, nbr_idx, pk_b1 + s * pk_elems(KTAP,32,64),
                       b_b1 + s * 64, x, 128, 64, vb, nullptr, 128, 64, 1, zp);
    cur = t[3];
  }

  // final: d_out[N,128] f32 = sparse_conv(cur, W_out) + b_out
  spconv<128,128,KTAP>(stream, X[cur], nbr_idx, pk_out, b_out,
                       nullptr, 0, 0, nullptr, (float*)d_out, 128, 0, 4, zp);
}
